// BuildifyMOE_6657199308969
// MI455X (gfx1250) — matrix operand order repaired and checked by reading, not compiled, not run on hardware
//
#include <hip/hip_runtime.h>
#include <math.h>

// ---------------------------------------------------------------------------
// MI455X (gfx1250, wave32) implementation.
// All large matmuls go through v_wmma_f32_16x16x32_f16 (f16 operands, f32 acc).
// All GEMM shapes are padded so the hot loop has ZERO guards/branches:
//   M = 1024 (mult of 128), K in {64,256,512,2048} (mult of 64),
//   N padded to mult of 64 (gate: 8->64, heads: 23->64) with zero columns.
// ---------------------------------------------------------------------------

typedef __attribute__((ext_vector_type(16))) _Float16 v16h;
typedef __attribute__((ext_vector_type(8)))  _Float16 half8;
typedef __attribute__((ext_vector_type(2)))  _Float16 half2;
typedef __attribute__((ext_vector_type(8)))  float    v8f;

__device__ __forceinline__ float gelu_f(float x) {
    // exact erf GELU (matches torch nn.GELU default / jax approximate=False)
    return 0.5f * x * (1.f + erff(x * 0.70710678118654752f));
}
__device__ __forceinline__ float sigmoid_f(float x) {
    return 1.f / (1.f + __expf(-x));
}

// ---------------------------------------------------------------------------
// WMMA GEMM:  C[M,Nout] = A[M,K] (f16 row-major) * B[K,Npad] (f16 row-major)
// Block tile 128x64, 8 waves (4x2), each wave 32x32 (2x2 WMMA tiles), BK=64
// (two WMMA K-steps per LDS stage -> 8 v_wmma per barrier pair).
// emode: 0 = store (+bias)
//        1 = gelu(x+bias) store
//        2 = atomicAdd(C, rowScale[r]*(x+bias))      (MoE combine)
//        4 = heads: sigmoid on cols 0-3,20,21; store to d_out rooms
// C (f32) and Ch (f16 copy) are each optional (null = skip).
// Requires: M % 128 == 0, K % 64 == 0, Npad % 64 == 0.
// ---------------------------------------------------------------------------
__global__ __launch_bounds__(256) void wmma_gemm_kernel(
    const _Float16* __restrict__ A, const _Float16* __restrict__ Bm,
    const float* __restrict__ bias,
    float* __restrict__ C, _Float16* __restrict__ Ch,
    const float* __restrict__ rowScale,
    int M, int Npad, int Nout, int K, int ldc, int emode)
{
    constexpr int BM = 128, BN = 64, BK = 64;
    __shared__ _Float16 sA[BM * BK];   // [m][k], 16 KB
    __shared__ _Float16 sB[BN * BK];   // [n][k] (transposed), 8 KB

    const int tid  = threadIdx.x;
    const int lane = tid & 31;
    const int wave = tid >> 5;
    const int wm = (wave & 3) * 32;    // wave row offset in block tile
    const int wn = (wave >> 2) * 32;   // wave col offset in block tile
    const int rowBlk = blockIdx.y * BM;
    const int colBlk = blockIdx.x * BN;

    v8f acc[2][2] = {};

    // cooperative-load coordinates (no bounds checks by construction)
    const int aRow = tid >> 1;          // 0..127
    const int aOff = (tid & 1) * 32;    // 0 or 32 halves
    const int bK0  = (tid >> 3) * 2;    // 0,2,..,62  (two K rows per thread)
    const int bN0  = (tid & 7) * 8;     // 0..56

    const _Float16* gaBase = A  + (size_t)(rowBlk + aRow) * K + aOff;
    const _Float16* gbBase = Bm + (size_t)bK0 * Npad + colBlk + bN0;

    for (int kt = 0; kt < K; kt += BK) {
        // ---- issue ALL global loads first (distinct regs), then LDS stores ----
        const _Float16* ga = gaBase + kt;
        const _Float16* gb = gbBase + (size_t)kt * Npad;
        half8 a0 = *(const half8*)(ga);
        half8 a1 = *(const half8*)(ga + 8);
        half8 a2 = *(const half8*)(ga + 16);
        half8 a3 = *(const half8*)(ga + 24);
        half8 r0 = *(const half8*)(gb);
        half8 r1 = *(const half8*)(gb + Npad);
        if (kt + BK < K) {  // global_prefetch for next A tile
            __builtin_prefetch(gaBase + kt + BK, 0, 1);
        }
        *(half8*)(sA + aRow * BK + aOff)      = a0;
        *(half8*)(sA + aRow * BK + aOff + 8)  = a1;
        *(half8*)(sA + aRow * BK + aOff + 16) = a2;
        *(half8*)(sA + aRow * BK + aOff + 24) = a3;
#pragma unroll
        for (int q = 0; q < 8; ++q) {           // B transposed, packed K pairs
            half2 p; p[0] = r0[q]; p[1] = r1[q];
            *(half2*)(sB + (bN0 + q) * BK + bK0) = p;
        }
        __syncthreads();

        // ---- two WMMA K-steps per stage ----
#pragma unroll
        for (int kk = 0; kk < 2; ++kk) {
            v16h af[2], bf[2];
#pragma unroll
            for (int i = 0; i < 2; ++i) {
                // A 16x32 frag: lanes 0-15 M=lane K=0..7,16..23; lanes 16-31 K=8..15,24..31
                int m  = wm + i * 16 + (lane & 15);
                int kh = kk * 32 + (lane >> 4) * 8;
                half8 lo = *(const half8*)(sA + m * BK + kh);
                half8 hi = *(const half8*)(sA + m * BK + kh + 16);
#pragma unroll
                for (int q = 0; q < 8; ++q) { af[i][q] = lo[q]; af[i][q + 8] = hi[q]; }
            }
#pragma unroll
            for (int j = 0; j < 2; ++j) {
                // B 32x16 frag: lanes 0-15 N=lane K=0..7,16..23; lanes 16-31 K=8..15,24..31
                int n  = wn + j * 16 + (lane & 15);
                int ks = kk * 32 + (lane >> 4) * 8;
                half8 lo = *(const half8*)(sB + n * BK + ks);
                half8 hi = *(const half8*)(sB + n * BK + ks + 16);
#pragma unroll
                for (int q = 0; q < 8; ++q) { bf[j][q] = lo[q]; bf[j][q + 8] = hi[q]; }
            }
#pragma unroll
            for (int i = 0; i < 2; ++i)
#pragma unroll
                for (int j = 0; j < 2; ++j)
                    acc[i][j] = __builtin_amdgcn_wmma_f32_16x16x32_f16(
                        false, af[i], false, bf[j], (short)0, acc[i][j], false, false);
        }
        __syncthreads();
    }

    // ---- epilogue (C/D layout: lane&15 -> col, (lane>>4)*8+v -> row) ----
#pragma unroll
    for (int i = 0; i < 2; ++i) {
#pragma unroll
        for (int j = 0; j < 2; ++j) {
            int r0i = rowBlk + wm + i * 16 + (lane >> 4) * 8;
            int c   = colBlk + wn + j * 16 + (lane & 15);
            if (c >= Nout) continue;
            float bv = bias ? bias[c] : 0.f;
#pragma unroll
            for (int v = 0; v < 8; ++v) {
                int r = r0i + v;
                float x = acc[i][j][v] + bv;
                if (emode == 1) x = gelu_f(x);
                if (emode == 4 && (c < 4 || c == 20 || c == 21)) x = sigmoid_f(x);
                if (emode == 2) {
                    atomicAdd(C + (size_t)r * ldc + c, rowScale[r] * x);
                } else {
                    if (C)  C[(size_t)r * ldc + c]  = x;
                    if (Ch) Ch[(size_t)r * ldc + c] = (_Float16)x;
                }
            }
        }
    }
}

// ---------------------------------------------------------------------------
// f32 -> f16 conversion
// ---------------------------------------------------------------------------
__global__ void cvt_f32_f16_kernel(const float* __restrict__ s,
                                   _Float16* __restrict__ d, int n)
{
    for (int i = blockIdx.x * blockDim.x + threadIdx.x; i < n;
         i += gridDim.x * blockDim.x)
        d[i] = (_Float16)s[i];
}

// f32 (K x N) -> f16 (K x Npad) with zero-filled pad columns
__global__ void cvt_pad_f16_kernel(const float* __restrict__ s,
                                   _Float16* __restrict__ d,
                                   int K, int N, int Npad)
{
    int i = blockIdx.x * blockDim.x + threadIdx.x;
    if (i >= K * Npad) return;
    int k = i / Npad, n = i % Npad;
    d[i] = (n < N) ? (_Float16)s[(size_t)k * N + n] : (_Float16)0.f;
}

// ---------------------------------------------------------------------------
// LayerNorm: out = LN(x [+ res]) * g + b ; optional gelu; optional "+ addAfter"
// One block per row, D == 512 (2 elems/thread), writes f32 and/or f16.
// ---------------------------------------------------------------------------
__global__ __launch_bounds__(256) void add_ln_kernel(
    const float* __restrict__ x, const float* __restrict__ res,
    const float* __restrict__ gamma, const float* __restrict__ beta,
    const float* __restrict__ addAfter,
    float* __restrict__ outF, _Float16* __restrict__ outH,
    int D, int doGelu)
{
    const int b = blockIdx.x;
    const int t = threadIdx.x;
    const size_t base = (size_t)b * D;
    float v0 = x[base + t]       + (res ? res[base + t]       : 0.f);
    float v1 = x[base + t + 256] + (res ? res[base + t + 256] : 0.f);

    __shared__ float sbuf[256];
    sbuf[t] = v0 + v1;
    __syncthreads();
    for (int o = 128; o > 0; o >>= 1) { if (t < o) sbuf[t] += sbuf[t + o]; __syncthreads(); }
    float mean = sbuf[0] / (float)D;
    __syncthreads();
    float d0 = v0 - mean, d1 = v1 - mean;
    sbuf[t] = d0 * d0 + d1 * d1;
    __syncthreads();
    for (int o = 128; o > 0; o >>= 1) { if (t < o) sbuf[t] += sbuf[t + o]; __syncthreads(); }
    float inv = rsqrtf(sbuf[0] / (float)D + 1e-5f);

    int cs[2] = { t, t + 256 };
    float ds[2] = { d0, d1 };
#pragma unroll
    for (int q = 0; q < 2; ++q) {
        int c = cs[q];
        float y = ds[q] * inv * gamma[c] + beta[c];
        if (doGelu) y = gelu_f(y);
        if (addAfter) y += addAfter[base + c];
        if (outF) outF[base + c] = y;
        if (outH) outH[base + c] = (_Float16)y;
    }
}

// ---------------------------------------------------------------------------
// Gating: softmax over E=8 logits -> gate_probs (to d_out), usage (atomic),
// top-2 softmax weights -> cwT[e][b] dense combine matrix.
// ---------------------------------------------------------------------------
__global__ void gating_kernel(const float* __restrict__ logits,
                              float* __restrict__ gate_probs,
                              float* __restrict__ cwT,
                              float* __restrict__ usage, int Bn)
{
    int b = blockIdx.x * blockDim.x + threadIdx.x;
    if (b >= Bn) return;
    float l[8], p[8];
    float mx = -1e30f;
#pragma unroll
    for (int e = 0; e < 8; ++e) { l[e] = logits[b * 8 + e]; mx = fmaxf(mx, l[e]); }
    float den = 0.f;
#pragma unroll
    for (int e = 0; e < 8; ++e) { p[e] = __expf(l[e] - mx); den += p[e]; }
#pragma unroll
    for (int e = 0; e < 8; ++e) {
        p[e] /= den;
        gate_probs[b * 8 + e] = p[e];
        atomicAdd(&usage[e], p[e]);
    }
    int i1 = 0;
#pragma unroll
    for (int e = 1; e < 8; ++e) if (l[e] > l[i1]) i1 = e;
    int i2 = -1;
#pragma unroll
    for (int e = 0; e < 8; ++e)
        if (e != i1 && (i2 < 0 || l[e] > l[i2])) i2 = e;
    float eb = __expf(l[i2] - l[i1]);
    float w1 = 1.f / (1.f + eb), w2 = eb * w1;
#pragma unroll
    for (int e = 0; e < 8; ++e)
        cwT[(size_t)e * Bn + b] = (e == i1) ? w1 : ((e == i2) ? w2 : 0.f);
}

__global__ void aux_loss_kernel(const float* __restrict__ usage,
                                float* __restrict__ out, int Bn, int En)
{
    if (threadIdx.x == 0 && blockIdx.x == 0) {
        float s = 0.f;
        for (int e = 0; e < En; ++e) {
            float u = usage[e] / (float)Bn - 1.f / (float)En;
            s += u * u;
        }
        out[0] = s;  // mean((u-1/E)^2)*E == sum
    }
}

// ---------------------------------------------------------------------------
// GRU input projection (K = RF = 7, too small for WMMA): gi = x_s @ wi + bi
// x_s = zeros for step 0 else teacher_rooms[:, step-1, :]
// ---------------------------------------------------------------------------
__global__ __launch_bounds__(256) void gru_input_kernel(
    const float* __restrict__ teacher, const float* __restrict__ wi,
    const float* __restrict__ bi, float* __restrict__ gi, int step)
{
    const int N3 = 3 * 512;
    int b = blockIdx.y;
    int j = blockIdx.x * blockDim.x + threadIdx.x;
    if (j >= N3) return;
    float acc = bi[j];
    if (step > 0) {
        const float* xr = teacher + ((size_t)b * 16 + (step - 1)) * 7;
#pragma unroll
        for (int r = 0; r < 7; ++r) acc += xr[r] * wi[(size_t)r * N3 + j];
    }
    gi[(size_t)b * N3 + j] = acc;
}

// ---------------------------------------------------------------------------
// GRU gate pointwise: h = (1-z)*n + z*h, in-place, writes f16 copy.
// ---------------------------------------------------------------------------
__global__ __launch_bounds__(256) void gru_pointwise_kernel(
    const float* __restrict__ gi, const float* __restrict__ gh,
    float* __restrict__ h, _Float16* __restrict__ hH, int Dd)
{
    int b = blockIdx.y;
    int d = blockIdx.x * blockDim.x + threadIdx.x;
    if (d >= Dd) return;
    const float* gib = gi + (size_t)b * 3 * Dd;
    const float* ghb = gh + (size_t)b * 3 * Dd;
    float r = sigmoid_f(gib[d] + ghb[d]);
    float z = sigmoid_f(gib[Dd + d] + ghb[Dd + d]);
    float n = tanhf(gib[2 * Dd + d] + r * ghb[2 * Dd + d]);
    float hv = h[(size_t)b * Dd + d];
    float o = (1.f - z) * n + z * hv;
    h[(size_t)b * Dd + d]  = o;
    hH[(size_t)b * Dd + d] = (_Float16)o;
}

// ---------------------------------------------------------------------------
// Pack the 5 output heads into one f16 512x64 matrix (cols 23..63 zero) + bias
// ---------------------------------------------------------------------------
__global__ void pack_heads_kernel(
    const float* __restrict__ cw_, const float* __restrict__ tw,
    const float* __restrict__ zw,  const float* __restrict__ ew,
    const float* __restrict__ sw,
    const float* __restrict__ cb,  const float* __restrict__ tb,
    const float* __restrict__ zb,  const float* __restrict__ eb,
    const float* __restrict__ sb,
    _Float16* __restrict__ Wh, float* __restrict__ Bh)
{
    int i = blockIdx.x * blockDim.x + threadIdx.x;
    if (i >= 512 * 64) return;
    int k = i / 64, c = i % 64;
    float v = 0.f;
    if (c < 4)        v = cw_[k * 4 + c];
    else if (c < 20)  v = tw[k * 16 + (c - 4)];
    else if (c == 20) v = zw[k];
    else if (c == 21) v = ew[k];
    else if (c == 22) v = sw[k];
    Wh[i] = (_Float16)v;
    if (k == 0) {
        float bv = 0.f;
        if (c < 4)        bv = cb[c];
        else if (c < 20)  bv = tb[c - 4];
        else if (c == 20) bv = zb[0];
        else if (c == 21) bv = eb[0];
        else if (c == 22) bv = sb[0];
        Bh[c] = bv;
    }
}

// ---------------------------------------------------------------------------
// Host orchestration
// ---------------------------------------------------------------------------
extern "C" void kernel_launch(void* const* d_in, const int* in_sizes, int n_in,
                              void* d_out, int out_size, void* d_ws, size_t ws_size,
                              hipStream_t stream)
{
    constexpr int B = 1024, F_IN = 64, D = 512, HFF = 2048, E = 8, T = 16,
                  RF = 7, S = 16, Gd = 256;
    constexpr int OC = 4 + T + 3;  // 23 output channels

    // ---- inputs (setup_inputs dict order, params flattened in insertion order)
    const float* constraints = (const float*)d_in[0];
    const float* teacher     = (const float*)d_in[1];
    const float* proj_w  = (const float*)d_in[2];
    const float* proj_b  = (const float*)d_in[3];
    const float* ln0_g   = (const float*)d_in[4];
    const float* ln0_b   = (const float*)d_in[5];
    const float* attn_wv = (const float*)d_in[6];
    const float* attn_bv = (const float*)d_in[7];
    const float* attn_wo = (const float*)d_in[8];
    const float* attn_bo = (const float*)d_in[9];
    const float* ln1_g   = (const float*)d_in[10];
    const float* ln1_b   = (const float*)d_in[11];
    const float* ff_w1   = (const float*)d_in[12];
    const float* ff_b1   = (const float*)d_in[13];
    const float* ff_w2   = (const float*)d_in[14];
    const float* ff_b2   = (const float*)d_in[15];
    const float* ln2_g   = (const float*)d_in[16];
    const float* ln2_b   = (const float*)d_in[17];
    const float* gate_w1 = (const float*)d_in[18];
    const float* gate_b1 = (const float*)d_in[19];
    const float* gate_w2 = (const float*)d_in[20];
    const float* gate_b2 = (const float*)d_in[21];
    const float* exp_w1  = (const float*)d_in[22];
    const float* exp_b1  = (const float*)d_in[23];
    const float* exp_w2  = (const float*)d_in[24];
    const float* exp_b2  = (const float*)d_in[25];
    const float* mix_w   = (const float*)d_in[26];
    const float* mix_b   = (const float*)d_in[27];
    const float* mix_ln_g = (const float*)d_in[28];
    const float* mix_ln_b = (const float*)d_in[29];
    const float* gru_wi0 = (const float*)d_in[30];
    const float* gru_wh0 = (const float*)d_in[31];
    const float* gru_bi0 = (const float*)d_in[32];
    const float* gru_bh0 = (const float*)d_in[33];
    const float* gru_wi1 = (const float*)d_in[34];
    const float* gru_wh1 = (const float*)d_in[35];
    const float* gru_bi1 = (const float*)d_in[36];
    const float* gru_bh1 = (const float*)d_in[37];
    const float* comb_w  = (const float*)d_in[38];
    const float* comb_b  = (const float*)d_in[39];
    const float* comb_ln_g = (const float*)d_in[40];
    const float* comb_ln_b = (const float*)d_in[41];
    const float* coord_w = (const float*)d_in[42];
    const float* coord_b = (const float*)d_in[43];
    const float* type_w  = (const float*)d_in[44];
    const float* type_b  = (const float*)d_in[45];
    const float* zone_w  = (const float*)d_in[46];
    const float* zone_b  = (const float*)d_in[47];
    const float* ext_w   = (const float*)d_in[48];
    const float* ext_b   = (const float*)d_in[49];
    const float* stop_w  = (const float*)d_in[50];
    const float* stop_b  = (const float*)d_in[51];
    (void)in_sizes; (void)n_in; (void)out_size; (void)ws_size;

    float* out = (float*)d_out;
    float* out_rooms = out;                       // B*S*23
    float* out_gate  = out + (size_t)B * S * OC;  // B*E
    float* out_aux   = out_gate + (size_t)B * E;  // 1

    // ---- workspace carving (256B aligned) ----
    uintptr_t base = (uintptr_t)d_ws;
    size_t off = 0;
    auto allocB = [&](size_t bytes) -> void* {
        off = (off + 255) & ~(size_t)255;
        void* p = (void*)(base + off);
        off += bytes;
        return p;
    };
    auto allocH = [&](size_t n) { return (_Float16*)allocB(n * sizeof(_Float16)); };
    auto allocF = [&](size_t n) { return (float*)allocB(n * sizeof(float)); };

    // f16 weights (N padded to multiples of 64 where needed)
    _Float16* c_h     = allocH((size_t)B * F_IN);
    _Float16* projW_h = allocH((size_t)F_IN * D);
    _Float16* wv_h    = allocH((size_t)D * D);
    _Float16* wo_h    = allocH((size_t)D * D);
    _Float16* ffw1_h  = allocH((size_t)D * HFF);
    _Float16* ffw2_h  = allocH((size_t)HFF * D);
    _Float16* gw1_h   = allocH((size_t)D * Gd);
    _Float16* gw2_h   = allocH((size_t)Gd * 64);   // padded 8 -> 64
    _Float16* expw1_h = allocH((size_t)E * D * HFF);
    _Float16* expw2_h = allocH((size_t)E * HFF * D);
    _Float16* mixw_h  = allocH((size_t)D * D);
    _Float16* wh0_h   = allocH((size_t)D * 3 * D);
    _Float16* wi1_h   = allocH((size_t)D * 3 * D);
    _Float16* wh1_h   = allocH((size_t)D * 3 * D);
    _Float16* combA_h = allocH((size_t)D * D);   // comb_w rows 0..D-1  (context)
    _Float16* combB_h = allocH((size_t)D * D);   // comb_w rows D..2D-1 (h2)
    _Float16* headW_h = allocH((size_t)D * 64);  // packed heads, padded 23 -> 64
    float*    headB   = allocF(64);

    // activations
    float*    t0      = allocF((size_t)B * D);       // generic BxD f32
    float*    h_f     = allocF((size_t)B * D);
    _Float16* h_h     = allocH((size_t)B * D);
    _Float16* v_h     = allocH((size_t)B * D);
    float*    emb_f   = allocF((size_t)B * D);
    _Float16* emb_h   = allocH((size_t)B * D);
    _Float16* t1_h    = allocH((size_t)B * HFF);     // BxHFF f16 (ff / experts)
    _Float16* g1_h    = allocH((size_t)B * Gd);
    float*    logits  = allocF((size_t)B * E);
    float*    cwT     = allocF((size_t)E * B);
    float*    usage   = allocF(E);
    float*    mixed   = allocF((size_t)B * D);
    _Float16* mixed_h = allocH((size_t)B * D);
    _Float16* ctx_h   = allocH((size_t)B * D);
    float*    combA   = allocF((size_t)B * D);
    float*    h1_f    = allocF((size_t)B * D);
    float*    h2_f    = allocF((size_t)B * D);
    _Float16* h1_h    = allocH((size_t)B * D);
    _Float16* h2_h    = allocH((size_t)B * D);
    float*    gi_f    = allocF((size_t)B * 3 * D);
    float*    gh_f    = allocF((size_t)B * 3 * D);
    _Float16* hid_h   = allocH((size_t)B * D);

    // ---- launch helpers ----
    auto cvt = [&](const float* s, _Float16* d, int n) {
        int blocks = (n + 1023) / 1024;
        cvt_f32_f16_kernel<<<dim3(blocks), dim3(256), 0, stream>>>(s, d, n);
    };
    auto gemm = [&](const _Float16* A, const _Float16* Bw, const float* bias,
                    float* C, _Float16* Ch, const float* rs,
                    int M, int Npad, int Nout, int K, int ldc, int emode) {
        dim3 g(Npad / 64, M / 128);
        wmma_gemm_kernel<<<g, dim3(256), 0, stream>>>(A, Bw, bias, C, Ch, rs,
                                                      M, Npad, Nout, K, ldc, emode);
    };
    auto lnorm = [&](const float* x, const float* res, const float* g_,
                     const float* b_, const float* addAfter,
                     float* oF, _Float16* oH, int doGelu) {
        add_ln_kernel<<<dim3(B), dim3(256), 0, stream>>>(x, res, g_, b_, addAfter,
                                                         oF, oH, D, doGelu);
    };

    // ---- weight conversion ----
    cvt(constraints, c_h, B * F_IN);
    cvt(proj_w, projW_h, F_IN * D);
    cvt(attn_wv, wv_h, D * D);
    cvt(attn_wo, wo_h, D * D);
    cvt(ff_w1, ffw1_h, D * HFF);
    cvt(ff_w2, ffw2_h, HFF * D);
    cvt(gate_w1, gw1_h, D * Gd);
    cvt_pad_f16_kernel<<<dim3((Gd * 64 + 255) / 256), dim3(256), 0, stream>>>(
        gate_w2, gw2_h, Gd, E, 64);
    cvt(exp_w1, expw1_h, E * D * HFF);
    cvt(exp_w2, expw2_h, E * HFF * D);
    cvt(mix_w, mixw_h, D * D);
    cvt(gru_wh0, wh0_h, D * 3 * D);
    cvt(gru_wi1, wi1_h, D * 3 * D);
    cvt(gru_wh1, wh1_h, D * 3 * D);
    cvt(comb_w, combA_h, D * D);                 // rows 0..511
    cvt(comb_w + (size_t)D * D, combB_h, D * D); // rows 512..1023
    pack_heads_kernel<<<dim3((512 * 64 + 255) / 256), dim3(256), 0, stream>>>(
        coord_w, type_w, zone_w, ext_w, stop_w,
        coord_b, type_b, zone_b, ext_b, stop_b, headW_h, headB);

    // ---- ConstraintEncoder ----
    gemm(c_h, projW_h, proj_b, t0, nullptr, nullptr, B, D, D, F_IN, D, 0);
    lnorm(t0, nullptr, ln0_g, ln0_b, nullptr, h_f, h_h, 0);
    gemm(h_h, wv_h, attn_bv, nullptr, v_h, nullptr, B, D, D, D, D, 0);
    gemm(v_h, wo_h, attn_bo, t0, nullptr, nullptr, B, D, D, D, D, 0);
    lnorm(t0, h_f, ln1_g, ln1_b, nullptr, h_f, h_h, 0);
    gemm(h_h, ffw1_h, ff_b1, nullptr, t1_h, nullptr, B, HFF, HFF, D, HFF, 1);
    gemm(t1_h, ffw2_h, ff_b2, t0, nullptr, nullptr, B, D, D, HFF, D, 0);
    lnorm(t0, h_f, ln2_g, ln2_b, nullptr, emb_f, emb_h, 0);

    // ---- SparseGating ----
    gemm(emb_h, gw1_h, gate_b1, nullptr, g1_h, nullptr, B, Gd, Gd, D, Gd, 1);
    gemm(g1_h, gw2_h, gate_b2, logits, nullptr, nullptr, B, 64, E, Gd, E, 0);
    hipMemsetAsync(usage, 0, E * sizeof(float), stream);
    hipMemsetAsync(mixed, 0, (size_t)B * D * sizeof(float), stream);
    gating_kernel<<<dim3((B + 255) / 256), dim3(256), 0, stream>>>(
        logits, out_gate, cwT, usage, B);
    aux_loss_kernel<<<dim3(1), dim3(32), 0, stream>>>(usage, out_aux, B, E);

    // ---- Experts (dense compute, sparse scaled atomic combine into mixed) ----
    for (int e = 0; e < E; ++e) {
        gemm(emb_h, expw1_h + (size_t)e * D * HFF, exp_b1 + (size_t)e * HFF,
             nullptr, t1_h, nullptr, B, HFF, HFF, D, HFF, 1);              // gelu
        gemm(t1_h, expw2_h + (size_t)e * HFF * D, exp_b2 + (size_t)e * D,
             mixed, nullptr, cwT + (size_t)e * B, B, D, D, HFF, D, 2);     // atomic
    }

    // ---- mixture projection: context = gelu(LN(mixed@W+b)) + emb ----
    cvt(mixed, mixed_h, B * D);
    gemm(mixed_h, mixw_h, mix_b, t0, nullptr, nullptr, B, D, D, D, D, 0);
    lnorm(t0, nullptr, mix_ln_g, mix_ln_b, emb_f, nullptr, ctx_h, 1);

    // ---- decoder-invariant: combA = context @ comb_w[:D] + comb_b ----
    gemm(ctx_h, combA_h, comb_b, combA, nullptr, nullptr, B, D, D, D, D, 0);

    // ---- GRU init ----
    hipMemsetAsync(h1_f, 0, (size_t)B * D * sizeof(float), stream);
    hipMemsetAsync(h2_f, 0, (size_t)B * D * sizeof(float), stream);
    hipMemsetAsync(h1_h, 0, (size_t)B * D * sizeof(_Float16), stream);
    hipMemsetAsync(h2_h, 0, (size_t)B * D * sizeof(_Float16), stream);

    // ---- RoomDecoder: 16 teacher-forced steps ----
    for (int s = 0; s < S; ++s) {
        gru_input_kernel<<<dim3(6, B), dim3(256), 0, stream>>>(
            teacher, gru_wi0, gru_bi0, gi_f, s);
        gemm(h1_h, wh0_h, gru_bh0, gh_f, nullptr, nullptr, B, 3 * D, 3 * D, D, 3 * D, 0);
        gru_pointwise_kernel<<<dim3(2, B), dim3(256), 0, stream>>>(
            gi_f, gh_f, h1_f, h1_h, D);

        gemm(h1_h, wi1_h, gru_bi1, gi_f, nullptr, nullptr, B, 3 * D, 3 * D, D, 3 * D, 0);
        gemm(h2_h, wh1_h, gru_bh1, gh_f, nullptr, nullptr, B, 3 * D, 3 * D, D, 3 * D, 0);
        gru_pointwise_kernel<<<dim3(2, B), dim3(256), 0, stream>>>(
            gi_f, gh_f, h2_f, h2_h, D);

        // hid = gelu(LN(combA + h2@comb_w[D:]))
        gemm(h2_h, combB_h, nullptr, t0, nullptr, nullptr, B, D, D, D, D, 0);
        lnorm(t0, combA, comb_ln_g, comb_ln_b, nullptr, nullptr, hid_h, 1);

        // heads -> rooms[:, s, :] (per-column sigmoid fused in epilogue)
        gemm(hid_h, headW_h, headB, out_rooms + (size_t)s * OC, nullptr, nullptr,
             B, 64, OC, D, S * OC, 4);
    }
}
